// Flow_FEDformer_80556406604656
// MI455X (gfx1250) — matrix operand order repaired and checked by reading, not compiled, not run on hardware
//
#include <hip/hip_runtime.h>

// FEDformer frequency block on MI455X (gfx1250, wave32, WMMA).
// B=32 H=8 L=4096 E=64 MODES=64.

#define L_LEN   4096
#define E_DIM   64
#define NMODES  64
#define BH_CNT  256   // B*H

typedef __attribute__((ext_vector_type(16))) __bf16 v16bf;
typedef __attribute__((ext_vector_type(8)))  float  v8f;
typedef __attribute__((ext_vector_type(4)))  float  f32x4;

union Frag16 { uint4 u[2]; v16bf v; };

// K-position inside a 16x32 bf16 A-tile for (lane, j): contiguous order,
// lane half khalf = lane>>4 holds K = khalf*16 + 0..15 (matches B reads).
// kl = khalf*16 + j.
__device__ __forceinline__ int a_klocal(int lane, int j) {
    return (((lane >> 4) & 1) << 4) + j;
}

// ---------------------------------------------------------------------------
// Kernel 0a: build pre-swizzled bf16 A-fragment bases.
//   Forward:  Fr[k,l] =  cos(2*pi*idx[k]*l/L),  Fi[k,l] = -sin(...)
//             tiles: (Mtile over modes: 4) x (Ktile over l: 128), 512 bf16/tile
//   Inverse:  Cr[l,k] =  s_k*cos(2*pi*idx[k]*l/L), Ci[l,k] = -s_k*sin(...)
//             tiles: (Mtile over l: 256) x (Ktile over modes: 2)
// Tile storage: lane-major [32 lanes][16 halfs] -> one 32B load per lane.
// ---------------------------------------------------------------------------
__global__ __launch_bounds__(256) void basis_kernel(
    const int* __restrict__ index,
    __bf16* __restrict__ Fr, __bf16* __restrict__ Fi,
    __bf16* __restrict__ Cr, __bf16* __restrict__ Ci) {
    const float W0 = 6.283185307179586f / (float)L_LEN;
    int tid  = blockIdx.x * 256 + threadIdx.x;
    int tile = tid >> 5;
    int lane = tid & 31;
    if (tile < 1024) {               // forward basis
        int mat   = tile >> 9;       // 0 = Re, 1 = Im
        int t     = tile & 511;      // Mtile*128 + Ktile
        int Mtile = t >> 7;
        int Ktile = t & 127;
        int m     = Mtile * 16 + (lane & 15);
        long long km = (long long)index[m];
        __bf16* dst = (mat ? Fi : Fr) + (size_t)t * 512 + lane * 16;
#pragma unroll
        for (int j = 0; j < 16; ++j) {
            int l  = Ktile * 32 + a_klocal(lane, j);
            int p  = (int)((km * (long long)l) & (L_LEN - 1));  // exact phase mod L
            float th = (float)p * W0;
            float s, c; __sincosf(th, &s, &c);
            dst[j] = (__bf16)(mat ? -s : c);
        }
    } else {                         // inverse basis
        tile -= 1024;
        int mat   = tile >> 9;       // 0 = Re, 1 = Im
        int t     = tile & 511;      // Mtile*2 + Ktile
        int Mtile = t >> 1;
        int Ktile = t & 1;
        int l     = Mtile * 16 + (lane & 15);
        __bf16* dst = (mat ? Ci : Cr) + (size_t)t * 512 + lane * 16;
#pragma unroll
        for (int j = 0; j < 16; ++j) {
            int mo = Ktile * 32 + a_klocal(lane, j);
            long long km = (long long)index[mo];
            float sc = (km == 0) ? (1.0f / L_LEN) : (2.0f / L_LEN);
            int p  = (int)((km * (long long)l) & (L_LEN - 1));
            float th = (float)p * W0;
            float s, c; __sincosf(th, &s, &c);
            dst[j] = (__bf16)(mat ? (-s * sc) : (c * sc));
        }
    }
}

// ---------------------------------------------------------------------------
// Kernel 0b: one-time weights transpose so the mix stage reads coalesced.
//   wT[k][e][o][2] = w[e][o][k][2]      (2 MB, L2-resident afterwards)
// ---------------------------------------------------------------------------
__global__ __launch_bounds__(256) void wtrans_kernel(
    const float* __restrict__ w, float* __restrict__ wT) {
    int p = blockIdx.x * 256 + threadIdx.x;    // 262144 complex pairs
    int e = p >> 12;
    int o = (p >> 6) & 63;
    int k = p & 63;
    float2 v = ((const float2*)w)[p];          // coalesced read
    ((float2*)wT)[(k << 12) + (e << 6) + o] = v;
}

// ---------------------------------------------------------------------------
// Kernel 1: sel[bh, ri, k, e] = sum_l F[ri][k,l] * x[bh, l, e]
// One block per (b,h). 8 waves: wave = (Mtile<<1)|ri, each owns 4 N-tiles.
// K=4096 in 64 steps of 64; x chunk converted to bf16 and transposed into a
// double-buffered LDS tile (ds_store_b64 packs of 4), so the next chunk's
// non-temporal global loads overlap with 8 WMMAs/wave per barrier.
// ---------------------------------------------------------------------------
__global__ __launch_bounds__(256) void forward_kernel(
    const float* __restrict__ x,
    const __bf16* __restrict__ Fr, const __bf16* __restrict__ Fi,
    float* __restrict__ sel) {
    __shared__ __bf16 xT[2][E_DIM * 64];       // [buf][e][l], 2 x 8 KB
    int bh   = blockIdx.x;
    int tid  = threadIdx.x;
    int wave = tid >> 5, lane = tid & 31;
    int ri = wave & 1, Mtile = wave >> 1;
    int eloc = lane & 15, khalf = lane >> 4;
    int eg = tid & 15, lg = tid >> 4;          // staging: 4x4 f32 block per thread

    const __bf16* Asw = (ri ? Fi : Fr) + (size_t)Mtile * 128 * 512 + lane * 16;
    const f32x4*  xb  = (const f32x4*)(x + (size_t)bh * L_LEN * E_DIM);

    v8f acc[4] = {};
    f32x4 reg[4];

    // prefetch + stage chunk 0
#pragma unroll
    for (int r = 0; r < 4; ++r)
        reg[r] = __builtin_nontemporal_load(&xb[(size_t)(lg * 4 + r) * 16 + eg]);
#pragma unroll
    for (int e = 0; e < 4; ++e) {
        union { __bf16 h[4]; uint2 u; } pk;
#pragma unroll
        for (int r = 0; r < 4; ++r) pk.h[r] = (__bf16)reg[r][e];
        *(uint2*)&xT[0][(eg * 4 + e) * 64 + lg * 4] = pk.u;
    }
    __syncthreads();

    for (int kt = 0; kt < 64; ++kt) {
        int cur = kt & 1;
        if (kt + 1 < 64) {                     // issue next chunk's loads
#pragma unroll
            for (int r = 0; r < 4; ++r)
                reg[r] = __builtin_nontemporal_load(
                    &xb[(size_t)((kt + 1) * 64 + lg * 4 + r) * 16 + eg]);
        }
#pragma unroll
        for (int sub = 0; sub < 2; ++sub) {
            Frag16 a;
            const uint4* ap = (const uint4*)(Asw + (size_t)(kt * 2 + sub) * 512);
            a.u[0] = ap[0]; a.u[1] = ap[1];
#pragma unroll
            for (int n = 0; n < 4; ++n) {
                Frag16 b;
                const uint4* bp = (const uint4*)
                    (&xT[cur][(n * 16 + eloc) * 64 + sub * 32 + khalf * 16]);
                b.u[0] = bp[0]; b.u[1] = bp[1];
                acc[n] = __builtin_amdgcn_wmma_f32_16x16x32_bf16(
                    false, a.v, false, b.v, (short)0, acc[n], false, false);
            }
        }
        if (kt + 1 < 64) {                     // stage next chunk into other buffer
#pragma unroll
            for (int e = 0; e < 4; ++e) {
                union { __bf16 h[4]; uint2 u; } pk;
#pragma unroll
                for (int r = 0; r < 4; ++r) pk.h[r] = (__bf16)reg[r][e];
                *(uint2*)&xT[cur ^ 1][(eg * 4 + e) * 64 + lg * 4] = pk.u;
            }
        }
        __syncthreads();
    }

    float* so = sel + (size_t)(bh * 2 + ri) * NMODES * E_DIM;
#pragma unroll
    for (int n = 0; n < 4; ++n) {
#pragma unroll
        for (int r = 0; r < 8; ++r) {
            int k = Mtile * 16 + khalf * 8 + r;
            int e = n * 16 + eloc;
            so[k * E_DIM + e] = acc[n][r];
        }
    }
}

// ---------------------------------------------------------------------------
// Kernel 2: complex per-mode mixing  O[k,o] = sum_e sel[k,e] * wT[k][e][o]
// One block per (bh, k), 64 threads (one output channel each). Writes bf16.
// ---------------------------------------------------------------------------
__global__ __launch_bounds__(64) void mix_kernel(
    const float* __restrict__ sel, const float* __restrict__ wT,
    __bf16* __restrict__ mixed) {
    int bh = blockIdx.x >> 6;
    int k  = blockIdx.x & 63;
    int o  = threadIdx.x;
    __shared__ float sr[E_DIM], si[E_DIM];
    size_t sb = ((size_t)(bh * 2) * NMODES + k) * E_DIM;
    sr[o] = sel[sb + o];
    si[o] = sel[sb + (size_t)NMODES * E_DIM + o];
    __syncthreads();
    const float2* wk = (const float2*)wT + ((size_t)k << 12) + o;
    float orr = 0.f, oii = 0.f;
    for (int e = 0; e < E_DIM; ++e) {
        float2 wc = wk[e << 6];                // contiguous across lanes
        float a = sr[e], b = si[e];
        orr += a * wc.x - b * wc.y;
        oii += a * wc.y + b * wc.x;
    }
    size_t mb = ((size_t)(bh * 2) * NMODES + k) * E_DIM + o;
    mixed[mb] = (__bf16)orr;
    mixed[mb + (size_t)NMODES * E_DIM] = (__bf16)oii;
}

// ---------------------------------------------------------------------------
// Kernel 3: y[bh, l, e] = Cr[l,:] @ Or[:, e] + Ci[l,:] @ Oi[:, e]
// One block per (b,h). O staged transposed [e][k] in LDS once; each wave
// walks 32 M-tiles, 4 N-tiles, 4 chained WMMAs (K=64 re + K=64 im).
// y stores are non-temporal (write-once stream).
// ---------------------------------------------------------------------------
__global__ __launch_bounds__(256) void inverse_kernel(
    const __bf16* __restrict__ mixed,
    const __bf16* __restrict__ Cr, const __bf16* __restrict__ Ci,
    float* __restrict__ y) {
    __shared__ __bf16 OT[2][E_DIM * NMODES];   // [ri][e][k], 16 KB
    int bh  = blockIdx.x;
    int tid = threadIdx.x;
    const uint4* mv = (const uint4*)(mixed + (size_t)bh * 2 * NMODES * E_DIM);
#pragma unroll
    for (int i = 0; i < 4; ++i) {
        int q = i * 256 + tid;                 // 1024 uint4 = 8192 bf16
        union { uint4 u; __bf16 h[8]; } pk;
        pk.u = mv[q];
        int f   = q * 8;
        int rif = f >> 12;
        int rem = f & 4095;
        int k = rem >> 6, e = rem & 63;
#pragma unroll
        for (int j = 0; j < 8; ++j) OT[rif][(e + j) * NMODES + k] = pk.h[j];
    }
    __syncthreads();

    int wave = tid >> 5, lane = tid & 31;
    int eloc = lane & 15, khalf = lane >> 4;
    float* yb = y + (size_t)bh * L_LEN * E_DIM;

    for (int mi = 0; mi < 32; ++mi) {
        int m = wave * 32 + mi;                // l-tile 0..255
        Frag16 ar[2], ai[2];
#pragma unroll
        for (int t = 0; t < 2; ++t) {
            const uint4* p = (const uint4*)(Cr + ((size_t)m * 2 + t) * 512 + lane * 16);
            ar[t].u[0] = p[0]; ar[t].u[1] = p[1];
            const uint4* q = (const uint4*)(Ci + ((size_t)m * 2 + t) * 512 + lane * 16);
            ai[t].u[0] = q[0]; ai[t].u[1] = q[1];
        }
#pragma unroll
        for (int n = 0; n < 4; ++n) {
            v8f acc = {};
#pragma unroll
            for (int t = 0; t < 2; ++t) {
                Frag16 b;
                const uint4* bp = (const uint4*)
                    (&OT[0][(n * 16 + eloc) * NMODES + t * 32 + khalf * 16]);
                b.u[0] = bp[0]; b.u[1] = bp[1];
                acc = __builtin_amdgcn_wmma_f32_16x16x32_bf16(
                    false, ar[t].v, false, b.v, (short)0, acc, false, false);
                const uint4* cp = (const uint4*)
                    (&OT[1][(n * 16 + eloc) * NMODES + t * 32 + khalf * 16]);
                b.u[0] = cp[0]; b.u[1] = cp[1];
                acc = __builtin_amdgcn_wmma_f32_16x16x32_bf16(
                    false, ai[t].v, false, b.v, (short)0, acc, false, false);
            }
#pragma unroll
            for (int r = 0; r < 8; ++r) {
                int l = m * 16 + khalf * 8 + r;
                int e = n * 16 + eloc;
                __builtin_nontemporal_store(acc[r], &yb[(size_t)l * E_DIM + e]);
            }
        }
    }
}

// ---------------------------------------------------------------------------
extern "C" void kernel_launch(void* const* d_in, const int* in_sizes, int n_in,
                              void* d_out, int out_size, void* d_ws, size_t ws_size,
                              hipStream_t stream) {
    const float* x     = (const float*)d_in[0];
    const float* w     = (const float*)d_in[1];
    const int*   index = (const int*)d_in[2];
    float*       out   = (float*)d_out;

    // Workspace layout (bytes):
    //   Fr 512K | Fi 512K | Cr 512K | Ci 512K | wT 2M | sel 8M (f32) | mixed 4M (bf16)
    char* ws = (char*)d_ws;
    __bf16* Fr    = (__bf16*)(ws);
    __bf16* Fi    = (__bf16*)(ws + (size_t)512 * 1024);
    __bf16* Cr    = (__bf16*)(ws + (size_t)1024 * 1024);
    __bf16* Ci    = (__bf16*)(ws + (size_t)1536 * 1024);
    float*  wT    = (float*)(ws + (size_t)2 * 1024 * 1024);
    float*  sel   = (float*)(ws + (size_t)4 * 1024 * 1024);
    __bf16* mixed = (__bf16*)(ws + (size_t)12 * 1024 * 1024);

    basis_kernel<<<256, 256, 0, stream>>>(index, Fr, Fi, Cr, Ci);
    wtrans_kernel<<<1024, 256, 0, stream>>>(w, wT);
    forward_kernel<<<BH_CNT, 256, 0, stream>>>(x, Fr, Fi, sel);
    mix_kernel<<<BH_CNT * NMODES, 64, 0, stream>>>(sel, wT, mixed);
    inverse_kernel<<<BH_CNT, 256, 0, stream>>>(mixed, Cr, Ci, out);
}
